// MultiHeadAttention_39273180955078
// MI455X (gfx1250) — matrix operand order repaired and checked by reading, not compiled, not run on hardware
//
#include <hip/hip_runtime.h>

// ---------------------------------------------------------------------------
// MI455X (gfx1250) MHA: bf16 WMMA GEMM pipeline with TDM (tensor_load_to_lds)
// double-buffered LDS staging for the TN GEMMs.
// ---------------------------------------------------------------------------

typedef __bf16 bf16;
typedef bf16  bf16x4  __attribute__((ext_vector_type(4)));
typedef bf16  bf16x8  __attribute__((ext_vector_type(8)));
typedef bf16  bf16x16 __attribute__((ext_vector_type(16)));
typedef float f32x4   __attribute__((ext_vector_type(4)));
typedef float f32x8   __attribute__((ext_vector_type(8)));
typedef unsigned int u32x4 __attribute__((ext_vector_type(4)));
typedef int   i32x4  __attribute__((ext_vector_type(4)));
typedef int   i32x8  __attribute__((ext_vector_type(8)));

#define BATCH   2
#define S_LEN   2048
#define DMODEL  1024
#define NH      16
#define DK      64

#define NEG_INF_F (-3.4028234663852886e38f)

#if defined(__has_builtin)
#if __has_builtin(__builtin_amdgcn_tensor_load_to_lds) && \
    __has_builtin(__builtin_amdgcn_s_wait_tensorcnt)
#define HAVE_TDM 1
#endif
#endif

static __device__ __forceinline__ bf16x16 cat8(bf16x8 lo, bf16x8 hi) {
  return __builtin_shufflevector(lo, hi, 0,1,2,3,4,5,6,7,8,9,10,11,12,13,14,15);
}

#ifdef HAVE_TDM
// ---------------------------------------------------------------------------
// Issue one TDM 2D tile load: rows x 32 bf16 tile, row stride `ld` elements,
// into LDS at byte offset lds_off with 40-bf16 (80B) padded row pitch.
//   D# group0: count=1, lds_addr, global_addr, type=2
//   D# group1: data_size=2B, pad_enable, pad_interval=16 DW, pad_amount=4 DW,
//              tensor_dim0=32, tensor_dim1=rows, tile_dim0=32, tile_dim1=rows,
//              tensor_dim0_stride=ld
// Toolchain note: amdgpu-toolchain clang-23 form takes 6 args
//   (u32x4 g0, i32x8 g1, i32x4 g2, i32x4 g3, i32x8 extra, i32 cpol).
// ---------------------------------------------------------------------------
static __device__ __forceinline__ void tdm_tile_load(unsigned lds_off,
                                                     const bf16* gptr,
                                                     int ld, int rows) {
  unsigned long long ga = (unsigned long long)(uintptr_t)gptr;
  u32x4 g0;
  g0[0] = 1u;                                        // count=1, user mode
  g0[1] = lds_off;                                   // lds_addr
  g0[2] = (unsigned)(ga & 0xffffffffu);              // global_addr[31:0]
  g0[3] = (unsigned)((ga >> 32) & 0x1ffffffu)        // global_addr[56:32]
        | (2u << 30);                                // type=2 (image)
  i32x8 g1;
  g1[0] = (int)((1u << 16)      // data_size = 1 -> 2 bytes
              | (1u << 20)      // pad_enable
              | (3u << 22)      // pad_interval: 3 -> 16 DWORDs (one 64B row)
              | (3u << 25));    // pad_amount:   3 -> 4 DWORDs (8 bf16)
  g1[1] = (int)(32u << 16);                 // tensor_dim0[15:0]=32 (atomic addr=0)
  g1[2] = (int)((unsigned)rows << 16);      // tensor_dim0 hi=0 | tensor_dim1 lo
  g1[3] = (int)(32u << 16);                 // tensor_dim1 hi=0 | tile_dim0=32
  g1[4] = rows;                             // tile_dim1 | tile_dim2=0
  g1[5] = ld;                               // tensor_dim0_stride[31:0]
  g1[6] = 0;                                // stride hi | dim1_stride lo
  g1[7] = 0;
  i32x4 z4 = {0, 0, 0, 0};                  // groups 2/3 unused (2D tensor)
  i32x8 z8 = {0, 0, 0, 0, 0, 0, 0, 0};
  __builtin_amdgcn_tensor_load_to_lds(g0, g1, z4, z4, z8, 0);
}
#endif

// ---------------------------------------------------------------------------
// f32 -> bf16 elementwise convert (4 elements / thread)
// ---------------------------------------------------------------------------
__global__ __launch_bounds__(256)
void cvt_bf16(const float* __restrict__ x, bf16* __restrict__ y, int n4) {
  int i = blockIdx.x * 256 + threadIdx.x;
  if (i < n4) {
    f32x4 v = ((const f32x4*)x)[i];
    bf16x4 o;
    o[0] = (bf16)v[0]; o[1] = (bf16)v[1]; o[2] = (bf16)v[2]; o[3] = (bf16)v[3];
    ((bf16x4*)y)[i] = o;
  }
}

// ---------------------------------------------------------------------------
// f32 [K=1024][N=1024] -> bf16 transposed [N][K]  (coalesced writes)
// ---------------------------------------------------------------------------
__global__ __launch_bounds__(256)
void cvt_bf16_t(const float* __restrict__ w, bf16* __restrict__ wt) {
  int o  = blockIdx.x * 256 + threadIdx.x;     // 0 .. 1M-1
  int nn = o >> 10;
  int kk = o & 1023;
  wt[o] = (bf16)w[(size_t)kk * DMODEL + nn];
}

// ---------------------------------------------------------------------------
// Generic TN GEMM:  C[M x N] = scale * A[M x K] * B[N x K]^T   (bf16 in, f32 acc)
// Batched over blockIdx.y = z;  b = z/heads, h = z%heads, per-operand strides.
// 256 threads = 8 waves; block tile 128x128x32; wave tile 32x64 (2x4 wmma).
// A/B tiles staged by the Tensor Data Mover, double-buffered in LDS.
// ---------------------------------------------------------------------------
template <bool OUT_BF16, bool CAUSAL>
__global__ __launch_bounds__(256)
void wmma_gemm_tn(const bf16* __restrict__ A, const bf16* __restrict__ B,
                  void* __restrict__ Cv,
                  int M, int N, int K, int lda, int ldb, int ldc,
                  long sAb, long sBb, long sCb,
                  int heads, long sAh, long sBh, long sCh,
                  float scale) {
  constexpr int BM = 128, BN = 128, BK = 32, SA = 40;  // 40*2B=80B row pitch
  __shared__ __align__(64) bf16 As[2][BM * SA];
  __shared__ __align__(64) bf16 Bs[2][BN * SA];

  const int tid  = threadIdx.x;
  const int wid  = tid >> 5;
  const int lane = tid & 31;
  const int half = lane >> 4;
  const int l15  = lane & 15;

  const int tiles_n = N / BN;
  const int m0 = (blockIdx.x / tiles_n) * BM;
  const int n0 = (blockIdx.x % tiles_n) * BN;

  const int z  = blockIdx.y;
  const int bb = z / heads;
  const int hh = z - bb * heads;
  const bf16* Az = A + bb * sAb + hh * sAh;
  const bf16* Bz = B + bb * sBb + hh * sBh;
  const long  c0 = bb * sCb + hh * sCh;

  const int wm = (wid & 3) * 32;   // 4 waves along M
  const int wn = (wid >> 2) * 64;  // 2 waves along N

  f32x8 acc[2][4] = {};

  const bf16* Atile = Az + (long)m0 * lda;  // k offset added per step
  const bf16* Btile = Bz + (long)n0 * ldb;
  const int nk = K / BK;

#ifdef HAVE_TDM
  const unsigned ldsA0 = (unsigned)(uintptr_t)&As[0][0];
  const unsigned ldsA1 = (unsigned)(uintptr_t)&As[1][0];
  const unsigned ldsB0 = (unsigned)(uintptr_t)&Bs[0][0];
  const unsigned ldsB1 = (unsigned)(uintptr_t)&Bs[1][0];
  if (tid < 32) {  // wave 0 drives the DMA
    tdm_tile_load(ldsA0, Atile, lda, BM);
    tdm_tile_load(ldsB0, Btile, ldb, BN);
  }
#else
  const int ar = tid >> 1;
  const int ah = (tid & 1) * 16;
#endif

  for (int kt = 0; kt < nk; ++kt) {
    const int cur = kt & 1;
    __syncthreads();  // everyone done reading buffer we are about to refill
#ifdef HAVE_TDM
    if (tid < 32) {
      if (kt + 1 < nk) {
        const int nxt = (kt + 1) & 1;
        tdm_tile_load(nxt ? ldsA1 : ldsA0, Atile + (kt + 1) * BK, lda, BM);
        tdm_tile_load(nxt ? ldsB1 : ldsB0, Btile + (kt + 1) * BK, ldb, BN);
        __builtin_amdgcn_s_wait_tensorcnt(2);  // current tile's 2 loads done
      } else {
        __builtin_amdgcn_s_wait_tensorcnt(0);
      }
    }
#else
    const int k0 = kt * BK;
    *(bf16x8*)&As[cur][ar * SA + ah] =
        *(const bf16x8*)&Atile[(long)ar * lda + k0 + ah];
    *(bf16x8*)&As[cur][ar * SA + ah + 8] =
        *(const bf16x8*)&Atile[(long)ar * lda + k0 + ah + 8];
    *(bf16x8*)&Bs[cur][ar * SA + ah] =
        *(const bf16x8*)&Btile[(long)ar * ldb + k0 + ah];
    *(bf16x8*)&Bs[cur][ar * SA + ah + 8] =
        *(const bf16x8*)&Btile[(long)ar * ldb + k0 + ah + 8];
#endif
    __syncthreads();  // tile `kt` visible to all waves

    bf16x16 afrag[2], bfrag[4];
#pragma unroll
    for (int i = 0; i < 2; ++i) {
      const bf16* pa = &As[cur][(wm + i * 16 + l15) * SA + 8 * half];
      afrag[i] = cat8(*(const bf16x8*)pa, *(const bf16x8*)(pa + 16));
    }
#pragma unroll
    for (int j = 0; j < 4; ++j) {
      const bf16* pb = &Bs[cur][(wn + j * 16 + l15) * SA + 8 * half];
      bfrag[j] = cat8(*(const bf16x8*)pb, *(const bf16x8*)(pb + 16));
    }

#pragma unroll
    for (int i = 0; i < 2; ++i)
#pragma unroll
      for (int j = 0; j < 4; ++j)
        acc[i][j] = __builtin_amdgcn_wmma_f32_16x16x32_bf16(
            false, afrag[i], false, bfrag[j], (short)0, acc[i][j], false, false);
  }

#pragma unroll
  for (int i = 0; i < 2; ++i) {
#pragma unroll
    for (int j = 0; j < 4; ++j) {
      const int gm = m0 + wm + i * 16 + 8 * half;  // + r
      const int gn = n0 + wn + j * 16 + l15;
#pragma unroll
      for (int r = 0; r < 8; ++r) {
        const int row = gm + r;
        float v = acc[i][j][r] * scale;
        if (CAUSAL && gn > row) v = NEG_INF_F;
        if constexpr (OUT_BF16)
          ((bf16*)Cv)[c0 + (long)row * ldc + gn] = (bf16)v;
        else
          ((float*)Cv)[c0 + (long)row * ldc + gn] = v;
      }
    }
  }
}

// ---------------------------------------------------------------------------
// Row softmax in place: one 256-thread block per row of 2048 f32.
// ---------------------------------------------------------------------------
__global__ __launch_bounds__(256)
void softmax_rows(float* __restrict__ attn) {
  __shared__ float red[256];
  float* p = attn + (long)blockIdx.x * S_LEN;
  const int t = threadIdx.x;

  float v[8];
  float m = NEG_INF_F;
#pragma unroll
  for (int i = 0; i < 8; ++i) {
    v[i] = p[t + i * 256];
    m = fmaxf(m, v[i]);
  }
  red[t] = m;
  __syncthreads();
  for (int s = 128; s > 0; s >>= 1) {
    if (t < s) red[t] = fmaxf(red[t], red[t + s]);
    __syncthreads();
  }
  m = red[0];
  __syncthreads();

  float sum = 0.f;
#pragma unroll
  for (int i = 0; i < 8; ++i) {
    v[i] = __expf(v[i] - m);
    sum += v[i];
  }
  red[t] = sum;
  __syncthreads();
  for (int s = 128; s > 0; s >>= 1) {
    if (t < s) red[t] += red[t + s];
    __syncthreads();
  }
  const float inv = 1.0f / red[0];
#pragma unroll
  for (int i = 0; i < 8; ++i) p[t + i * 256] = v[i] * inv;
}

// ---------------------------------------------------------------------------
// ctx = attn @ V per (b,h):  M=2048(q), N=64(d), K=2048(k).
// attn is f32 (converted to bf16 while staging); V tile transposed into LDS.
// Block: 128 q-rows; 8 waves, one 16-row stripe each, 4 wmma tiles along N.
// ---------------------------------------------------------------------------
__global__ __launch_bounds__(256)
void wmma_gemm_pv(const float* __restrict__ P, const bf16* __restrict__ V,
                  bf16* __restrict__ Ctx) {
  constexpr int BM = 128, BK = 32, SA = 40;
  __shared__ __align__(64) bf16 As[BM * SA];
  __shared__ __align__(64) bf16 Bs[DK * SA];

  const int tid  = threadIdx.x;
  const int wid  = tid >> 5;
  const int lane = tid & 31;
  const int half = lane >> 4;
  const int l15  = lane & 15;

  const int z  = blockIdx.y;
  const int bb = z >> 4;
  const int hh = z & 15;
  const float* Pz = P + (long)z * S_LEN * S_LEN;
  const bf16*  Vz = V + (long)bb * S_LEN * DMODEL + hh * DK;
  bf16*        Cz = Ctx + (long)bb * S_LEN * DMODEL + hh * DK;

  const int m0 = blockIdx.x * BM;

  f32x8 acc[4] = {};

  const int vk = tid >> 3;        // 0..31 : V row (k) staged by this thread
  const int vd = (tid & 7) * 8;   // 0..56 : V col chunk

  for (int k0 = 0; k0 < S_LEN; k0 += BK) {
    __syncthreads();
    // stage attn tile (f32 -> bf16): 128x32 = 512 chunks of 8
#pragma unroll
    for (int rep = 0; rep < 2; ++rep) {
      const int idx  = tid + rep * 256;
      const int arow = idx >> 2;
      const int ach  = (idx & 3) * 8;
      const float* src = &Pz[(long)(m0 + arow) * S_LEN + k0 + ach];
      f32x4 v0 = *(const f32x4*)src;
      f32x4 v1 = *(const f32x4*)(src + 4);
      bf16x8 bv;
      bv[0] = (bf16)v0[0]; bv[1] = (bf16)v0[1];
      bv[2] = (bf16)v0[2]; bv[3] = (bf16)v0[3];
      bv[4] = (bf16)v1[0]; bv[5] = (bf16)v1[1];
      bv[6] = (bf16)v1[2]; bv[7] = (bf16)v1[3];
      *(bf16x8*)&As[arow * SA + ach] = bv;
    }
    // stage V tile transposed: Bs[d][k] = V[k0+k][d]
    {
      bf16x8 bv = *(const bf16x8*)&Vz[(long)(k0 + vk) * DMODEL + vd];
#pragma unroll
      for (int i = 0; i < 8; ++i) Bs[(vd + i) * SA + vk] = bv[i];
    }
    __syncthreads();

    const bf16* pa = &As[(wid * 16 + l15) * SA + 8 * half];
    const bf16x16 afrag = cat8(*(const bf16x8*)pa, *(const bf16x8*)(pa + 16));
#pragma unroll
    for (int j = 0; j < 4; ++j) {
      const bf16* pb = &Bs[(j * 16 + l15) * SA + 8 * half];
      const bf16x16 bfrag = cat8(*(const bf16x8*)pb, *(const bf16x8*)(pb + 16));
      acc[j] = __builtin_amdgcn_wmma_f32_16x16x32_bf16(
          false, afrag, false, bfrag, (short)0, acc[j], false, false);
    }
  }

#pragma unroll
  for (int j = 0; j < 4; ++j) {
    const int gn = j * 16 + l15;
    const int gm = m0 + wid * 16 + 8 * half;
#pragma unroll
    for (int r = 0; r < 8; ++r)
      Cz[(long)(gm + r) * DMODEL + gn] = (bf16)acc[j][r];
  }
}

// ---------------------------------------------------------------------------
// Host-side orchestration
// ---------------------------------------------------------------------------
extern "C" void kernel_launch(void* const* d_in, const int* in_sizes, int n_in,
                              void* d_out, int out_size, void* d_ws, size_t ws_size,
                              hipStream_t stream) {
  const float* q_in = (const float*)d_in[0];
  const float* k_in = (const float*)d_in[1];
  const float* v_in = (const float*)d_in[2];
  // d_in[3] = mask (causal; applied analytically)
  const float* Wq = (const float*)d_in[4];
  const float* Wk = (const float*)d_in[5];
  const float* Wv = (const float*)d_in[6];
  const float* Wo = (const float*)d_in[7];

  float* out  = (float*)d_out;
  float* attn = out + (long)BATCH * S_LEN * DMODEL;  // 4,194,304 floats in

  const long NX = (long)BATCH * S_LEN * DMODEL;  // 4M activation elements
  const long NW = (long)DMODEL * DMODEL;         // 1M weight elements

  bf16* ws  = (bf16*)d_ws;
  bf16* Xq  = ws;
  bf16* Xk  = Xq + NX;
  bf16* Xv  = Xk + NX;
  bf16* Wqt = Xv + NX;
  bf16* Wkt = Wqt + NW;
  bf16* Wvt = Wkt + NW;
  bf16* Wot = Wvt + NW;
  bf16* Qb  = Wot + NW;
  bf16* Kb  = Qb + NX;
  bf16* Vb  = Kb + NX;
  bf16* Ctx = Vb + NX;

  const dim3 blk(256);

  // 1) convert activations + transposed weights to bf16
  cvt_bf16<<<(int)(NX / 4 / 256), blk, 0, stream>>>(q_in, Xq, (int)(NX / 4));
  cvt_bf16<<<(int)(NX / 4 / 256), blk, 0, stream>>>(k_in, Xk, (int)(NX / 4));
  cvt_bf16<<<(int)(NX / 4 / 256), blk, 0, stream>>>(v_in, Xv, (int)(NX / 4));
  cvt_bf16_t<<<(int)(NW / 256), blk, 0, stream>>>(Wq, Wqt);
  cvt_bf16_t<<<(int)(NW / 256), blk, 0, stream>>>(Wk, Wkt);
  cvt_bf16_t<<<(int)(NW / 256), blk, 0, stream>>>(Wv, Wvt);
  cvt_bf16_t<<<(int)(NW / 256), blk, 0, stream>>>(Wo, Wot);

  // 2) Q/K/V projections: [4096x1024] = X @ W  (bf16 out)
  const dim3 gproj((4096 / 128) * (1024 / 128), 1);
  wmma_gemm_tn<true, false><<<gproj, blk, 0, stream>>>(
      Xq, Wqt, Qb, 4096, 1024, 1024, DMODEL, DMODEL, DMODEL,
      0, 0, 0, 1, 0, 0, 0, 1.0f);
  wmma_gemm_tn<true, false><<<gproj, blk, 0, stream>>>(
      Xk, Wkt, Kb, 4096, 1024, 1024, DMODEL, DMODEL, DMODEL,
      0, 0, 0, 1, 0, 0, 0, 1.0f);
  wmma_gemm_tn<true, false><<<gproj, blk, 0, stream>>>(
      Xv, Wvt, Vb, 4096, 1024, 1024, DMODEL, DMODEL, DMODEL,
      0, 0, 0, 1, 0, 0, 0, 1.0f);

  // 3) scores = Q K^T / 8, causal-masked, straight into d_out attn region
  const dim3 gsc((S_LEN / 128) * (S_LEN / 128), BATCH * NH);
  wmma_gemm_tn<false, true><<<gsc, blk, 0, stream>>>(
      Qb, Kb, attn, S_LEN, S_LEN, DK, DMODEL, DMODEL, S_LEN,
      (long)S_LEN * DMODEL, (long)S_LEN * DMODEL, (long)NH * S_LEN * S_LEN,
      NH, DK, DK, (long)S_LEN * S_LEN, 0.125f);

  // 4) softmax rows in place
  softmax_rows<<<BATCH * NH * S_LEN, blk, 0, stream>>>(attn);

  // 5) ctx = attn @ V   (bf16 out, interleaved back into (b,s,d_model))
  const dim3 gpv(S_LEN / 128, BATCH * NH);
  wmma_gemm_pv<<<gpv, blk, 0, stream>>>(attn, Vb, Ctx);

  // 6) out = ctx @ W_o  (f32 out)
  wmma_gemm_tn<false, false><<<gproj, blk, 0, stream>>>(
      Ctx, Wot, out, 4096, 1024, 1024, DMODEL, DMODEL, DMODEL,
      0, 0, 0, 1, 0, 0, 0, 1.0f);
}
